// FusionRQVAE_v3_47923245089218
// MI455X (gfx1250) — compile-verified, matrix operand order checked by reading, not run on hardware
//
#include <hip/hip_runtime.h>
#include <stddef.h>

// ---------------------------------------------------------------------------
// CDNA5 (gfx1250) fusion RQ-VAE forward.
// Heavy GEMMs: v_wmma_f32_16x16x32_bf16 (wave32 WMMA), weights async-staged
// into LDS with global_load_async_to_lds_b128 / s_wait_asynccnt (ASYNCcnt).
// fp32 storage everywhere; bf16 only as WMMA input rounding.
// ---------------------------------------------------------------------------

typedef __bf16 v16bf __attribute__((ext_vector_type(16)));
typedef __bf16 v8bf  __attribute__((ext_vector_type(8)));
typedef float  v8f   __attribute__((ext_vector_type(8)));

#define NROWS 32768

// ---- async copy helpers (CDNA5 async-to-LDS path) -------------------------
__device__ __forceinline__ void async_g2l_b128(unsigned lds_off, const char* gaddr) {
  // per-lane 16B copy from global memory into LDS, tracked with ASYNCcnt (ISA 10.x / 15.18.3)
  asm volatile("global_load_async_to_lds_b128 %0, %1, off"
               :: "v"(lds_off), "v"(gaddr) : "memory");
}
__device__ __forceinline__ void wait_async0() {
  asm volatile("s_wait_asynccnt 0" ::: "memory");
}

// ---- WMMA fragment helpers -------------------------------------------------
// A fragment (16x32 bf16, MxK): lane L<16 -> row M=L, K in {kc+0..7, kc+16..23};
// lane L>=16 -> row M=L-16, K in {kc+8..15, kc+24..31}. Element e<8 maps to the
// low run, e>=8 to the +16 run. (ISA 7.12.2 layout, wave32.)
__device__ __forceinline__ v16bf cvt_a_frag(const float* p_lo, const float* p_hi) {
  float4 f0 = *(const float4*)(p_lo);
  float4 f1 = *(const float4*)(p_lo + 4);
  float4 f2 = *(const float4*)(p_hi);
  float4 f3 = *(const float4*)(p_hi + 4);
  v16bf a;
  a[0]=(__bf16)f0.x;  a[1]=(__bf16)f0.y;  a[2]=(__bf16)f0.z;  a[3]=(__bf16)f0.w;
  a[4]=(__bf16)f1.x;  a[5]=(__bf16)f1.y;  a[6]=(__bf16)f1.z;  a[7]=(__bf16)f1.w;
  a[8]=(__bf16)f2.x;  a[9]=(__bf16)f2.y;  a[10]=(__bf16)f2.z; a[11]=(__bf16)f2.w;
  a[12]=(__bf16)f3.x; a[13]=(__bf16)f3.y; a[14]=(__bf16)f3.z; a[15]=(__bf16)f3.w;
  return a;
}

// B fragment (32x16 bf16, KxN), symmetric layout: lane's column fixed, two
// contiguous 8-element K runs (16-byte loads each) from a [N][K] packed array.
__device__ __forceinline__ v16bf ld_b_frag(const __bf16* p) {
  v8bf lo = *(const v8bf*)(p);
  v8bf hi = *(const v8bf*)(p + 16);
  v16bf b;
#pragma unroll
  for (int i = 0; i < 8; ++i) { b[i] = lo[i]; b[8 + i] = hi[i]; }
  return b;
}

// ---- weight pack: fp32 [K,N] -> bf16 [N,K] --------------------------------
__global__ void rqvae_pack_w_kernel(const float* __restrict__ W,
                                    __bf16* __restrict__ Wt, int K, int Nc) {
  int idx = blockIdx.x * blockDim.x + threadIdx.x;
  if (idx >= K * Nc) return;
  int n = idx / K, k = idx % K;
  Wt[idx] = (__bf16)W[(size_t)k * Nc + n];
}

// ---- generic GEMM: C[M,Nc] = act(A[M,K] @ W[K,Nc] + bias) -----------------
// Block = 256 thr = 8 waves. Tile: 128 rows (16/wave) x 32 cols (2 WMMA tiles).
// The block's 32-column weight stripe (contiguous 32*K bf16 in packed Wt) is
// async-staged into dynamic LDS once and shared by all 8 waves.
__global__ __launch_bounds__(256) void rqvae_gemm_kernel(
    const float* __restrict__ A, const __bf16* __restrict__ Wt,
    const float* __restrict__ bias, float* __restrict__ C,
    int M, int K, int Nc, int relu) {
  (void)M;
  extern __shared__ __attribute__((aligned(16))) char smem[];
  __bf16* s_wt = (__bf16*)smem;  // [32][K] row-major

  const int col0 = blockIdx.x * 32;

  // async-stage weight stripe: contiguous 32*K*2 bytes
  {
    const char* gsrc = (const char*)(Wt + (size_t)col0 * K);
    const int nchunks = (K * 64) / 16;  // 32*K*2/16 = 4*K, multiple of 256
    for (int i = threadIdx.x; i < nchunks; i += 256)
      async_g2l_b128((unsigned)(size_t)(smem + (size_t)i * 16),
                     gsrc + (size_t)i * 16);
  }

  const int lane = threadIdx.x & 31;
  const int wave = threadIdx.x >> 5;
  const int row0 = blockIdx.y * 128 + wave * 16;
  const int mloc = lane & 15;
  const int koff = (lane < 16) ? 0 : 8;

  const float*  arow  = A + (size_t)(row0 + mloc) * K;
  const __bf16* b0row = s_wt + (size_t)mloc * K;
  const __bf16* b1row = s_wt + (size_t)(mloc + 16) * K;

  wait_async0();
  __syncthreads();

  v8f acc0 = {}; v8f acc1 = {};
  for (int kc = 0; kc < K; kc += 32) {
    if (kc + 32 < K) __builtin_prefetch(arow + kc + 32 + koff, 0, 3);
    v16bf a  = cvt_a_frag(arow + kc + koff, arow + kc + 16 + koff);
    v16bf b0 = ld_b_frag(b0row + kc + koff);
    v16bf b1 = ld_b_frag(b1row + kc + koff);
    acc0 = __builtin_amdgcn_wmma_f32_16x16x32_bf16(false, a, false, b0,
                                                   (short)0, acc0, false, false);
    acc1 = __builtin_amdgcn_wmma_f32_16x16x32_bf16(false, a, false, b1,
                                                   (short)0, acc1, false, false);
  }

  // C layout: element e -> (row0 + e + (lane<16?0:8), col0 + (lane&15))
  const int mbase = row0 + ((lane < 16) ? 0 : 8);
  const float bb0 = bias ? bias[col0 + mloc] : 0.0f;
  const float bb1 = bias ? bias[col0 + 16 + mloc] : 0.0f;
#pragma unroll
  for (int e = 0; e < 8; ++e) {
    float v0 = acc0[e] + bb0;
    float v1 = acc1[e] + bb1;
    if (relu) { v0 = fmaxf(v0, 0.0f); v1 = fmaxf(v1, 0.0f); }
    size_t r = (size_t)(mbase + e) * Nc;
    C[r + col0 + mloc]      = v0;
    C[r + col0 + 16 + mloc] = v1;
  }
}

// ---- cross-modal LoRA fusion ----------------------------------------------
// t_res = z_t + a*(z_v @ vfB) @ tfA ; v_res = z_v + a*(z_t @ tfB) @ vfA
// Also zeroes the x_q accumulators. One row per thread; factors in LDS.
__global__ __launch_bounds__(256) void rqvae_fusion_kernel(
    const float* __restrict__ z_t, const float* __restrict__ z_v,
    const float* __restrict__ tfA, const float* __restrict__ tfB,
    const float* __restrict__ vfA, const float* __restrict__ vfB,
    float* __restrict__ t_res, float* __restrict__ v_res,
    float* __restrict__ t_xq, float* __restrict__ v_xq) {
  __shared__ float s_tfA[2048], s_tfB[2048], s_vfA[2048], s_vfB[2048];
  for (int i = threadIdx.x; i < 2048; i += 256) {
    s_tfA[i] = tfA[i]; s_tfB[i] = tfB[i];
    s_vfA[i] = vfA[i]; s_vfB[i] = vfB[i];
  }
  __syncthreads();

  const size_t row = (size_t)blockIdx.x * 256 + threadIdx.x;
  const float* zt = z_t + row * 64;
  const float* zv = z_v + row * 64;
  float tmp[32];

  // delta_text = (zv @ vfB[64,32]) @ tfA[32,64]
#pragma unroll
  for (int r = 0; r < 32; ++r) tmp[r] = 0.0f;
  for (int k = 0; k < 64; ++k) {
    float z = zv[k];
#pragma unroll
    for (int r = 0; r < 32; ++r) tmp[r] += z * s_vfB[k * 32 + r];
  }
  for (int j = 0; j < 64; ++j) {
    float acc = 0.0f;
#pragma unroll
    for (int r = 0; r < 32; ++r) acc += tmp[r] * s_tfA[r * 64 + j];
    t_res[row * 64 + j] = zt[j] + 0.1f * acc;
    t_xq[row * 64 + j]  = 0.0f;
  }

  // delta_vis = (zt @ tfB[64,32]) @ vfA[32,64]
#pragma unroll
  for (int r = 0; r < 32; ++r) tmp[r] = 0.0f;
  for (int k = 0; k < 64; ++k) {
    float z = zt[k];
#pragma unroll
    for (int r = 0; r < 32; ++r) tmp[r] += z * s_tfB[k * 32 + r];
  }
  for (int j = 0; j < 64; ++j) {
    float acc = 0.0f;
#pragma unroll
    for (int r = 0; r < 32; ++r) acc += tmp[r] * s_vfA[r * 64 + j];
    v_res[row * 64 + j] = zv[j] + 0.1f * acc;
    v_xq[row * 64 + j]  = 0.0f;
  }
}

// ---- codebook build: cb = A[256,16] @ B[16,64], + bf16 copy + row norms ---
__global__ __launch_bounds__(256) void rqvae_codebook_kernel(
    const float* __restrict__ La, const float* __restrict__ Bm,
    float* __restrict__ cb_f32, __bf16* __restrict__ cb_b16,
    float* __restrict__ cb_norm) {
  __shared__ float sB[16 * 64];
  for (int i = threadIdx.x; i < 1024; i += 256) sB[i] = Bm[i];
  __syncthreads();
  const int code = threadIdx.x;  // 256 codes
  float a[16];
#pragma unroll
  for (int r = 0; r < 16; ++r) a[r] = La[code * 16 + r];
  float nrm = 0.0f;
  for (int j = 0; j < 64; ++j) {
    float acc = 0.0f;
#pragma unroll
    for (int r = 0; r < 16; ++r) acc += a[r] * sB[r * 64 + j];
    cb_f32[code * 64 + j] = acc;
    cb_b16[code * 64 + j] = (__bf16)acc;
    nrm += acc * acc;
  }
  cb_norm[code] = nrm;
}

// ---- fused RQ stage: WMMA distances + argmin + residual/x_q update --------
// Block = 256 thr = 8 waves, 128 rows/block. Codebook (256x64 bf16) is
// async-staged into LDS. argmin over ||cb||^2 - 2 r.cb (||r||^2 constant/row).
__global__ __launch_bounds__(256) void rqvae_rq_stage_kernel(
    float* __restrict__ residual, float* __restrict__ xq,
    const float* __restrict__ cb_f32, const __bf16* __restrict__ cb_b16,
    const float* __restrict__ cb_norm, float* __restrict__ idx_out,
    int stage, float* __restrict__ partial) {
  __shared__ __attribute__((aligned(16))) __bf16 s_cb[256 * 64];
  __shared__ float s_norm[256];
  __shared__ float s_part[8];

  const int tid = threadIdx.x;
  // async-stage codebook: 32KB = 2048 x 16B chunks
#pragma unroll
  for (int i = 0; i < 8; ++i) {
    int c = tid + i * 256;
    async_g2l_b128((unsigned)(size_t)((char*)s_cb + (size_t)c * 16),
                   (const char*)cb_b16 + (size_t)c * 16);
  }
  s_norm[tid] = cb_norm[tid];
  wait_async0();
  __syncthreads();

  const int lane = tid & 31;
  const int wave = tid >> 5;
  const int r0 = blockIdx.x * 128 + wave * 16;
  const int mloc = lane & 15;
  const int koff = (lane < 16) ? 0 : 8;

  // residual rows r0..r0+15 as two K=32 bf16 A-fragments
  const float* arow = residual + (size_t)(r0 + mloc) * 64;
  v16bf afrag[2];
#pragma unroll
  for (int ks = 0; ks < 2; ++ks)
    afrag[ks] = cvt_a_frag(arow + ks * 32 + koff, arow + ks * 32 + 16 + koff);

  float best_val[8]; int best_idx[8];
#pragma unroll
  for (int e = 0; e < 8; ++e) { best_val[e] = 3.4e38f; best_idx[e] = 0; }

  for (int t = 0; t < 16; ++t) {            // 16 tiles of 16 codes
    const int code = t * 16 + mloc;
    v8f acc = {};
#pragma unroll
    for (int ks = 0; ks < 2; ++ks) {
      v16bf b = ld_b_frag(s_cb + code * 64 + ks * 32 + koff);
      acc = __builtin_amdgcn_wmma_f32_16x16x32_bf16(false, afrag[ks], false, b,
                                                    (short)0, acc, false, false);
    }
    const float cn = s_norm[code];
#pragma unroll
    for (int e = 0; e < 8; ++e) {
      float v = cn - 2.0f * acc[e];
      int ci = code;
#pragma unroll
      for (int off = 8; off >= 1; off >>= 1) {  // min over 16 lanes (half-wave)
        float ov = __shfl_xor(v, off, 32);
        int   oi = __shfl_xor(ci, off, 32);
        if (ov < v || (ov == v && oi < ci)) { v = ov; ci = oi; }
      }
      if (v < best_val[e] || (v == best_val[e] && ci < best_idx[e])) {
        best_val[e] = v; best_idx[e] = ci;
      }
    }
  }

  // update: q = cb[idx]; residual -= q; xq += q; loss += ||q - r||^2
  float lsum = 0.0f;
#pragma unroll
  for (int m = 0; m < 16; ++m) {
    const int bi = __shfl(best_idx[m & 7], (m < 8) ? 0 : 16, 32);
    const float* q = cb_f32 + (size_t)bi * 64;
    const size_t base = (size_t)(r0 + m) * 64 + lane * 2;  // 2 cols per lane
    float rv0 = residual[base], rv1 = residual[base + 1];
    float q0 = q[lane * 2], q1 = q[lane * 2 + 1];
    residual[base]     = rv0 - q0;
    residual[base + 1] = rv1 - q1;
    xq[base]     += q0;
    xq[base + 1] += q1;
    float d0 = q0 - rv0, d1 = q1 - rv1;
    lsum += d0 * d0 + d1 * d1;
    if (lane == 0) idx_out[(size_t)(r0 + m) * 4 + stage] = (float)bi;
  }

  // deterministic block-level loss partial (no float atomics)
#pragma unroll
  for (int off = 16; off >= 1; off >>= 1) lsum += __shfl_xor(lsum, off, 32);
  if (lane == 0) s_part[wave] = lsum;
  __syncthreads();
  if (tid == 0) {
    float t = 0.0f;
    for (int w = 0; w < 8; ++w) t += s_part[w];
    partial[blockIdx.x] = t;
  }
}

// ---- finalize losses (fixed-order sum => deterministic) -------------------
__global__ void rqvae_finalize_kernel(const float* __restrict__ partial,
                                      float* __restrict__ out_t,
                                      float* __restrict__ out_v) {
  if (threadIdx.x == 0 && blockIdx.x == 0) {
    // loss = (1+beta) * mean_over_4_stages( sum_sq / (N*64) )
    const float scale = 1.25f / (4.0f * (float)NROWS * 64.0f);
    float st = 0.0f, sv = 0.0f;
    for (int i = 0; i < 1024; ++i) st += partial[i];          // text
    for (int i = 0; i < 1024; ++i) sv += partial[1024 + i];   // vis
    *out_t = st * scale;
    *out_v = sv * scale;
  }
}

// ---------------------------------------------------------------------------
extern "C" void kernel_launch(void* const* d_in, const int* in_sizes, int n_in,
                              void* d_out, int out_size, void* d_ws, size_t ws_size,
                              hipStream_t stream) {
  (void)in_sizes; (void)n_in; (void)out_size; (void)ws_size;
  const int N = NROWS;

  // -- inputs (recursive dict-insertion order of setup_inputs()) --
  const float* x_text = (const float*)d_in[0];   // [N,768]
  const float* x_vis  = (const float*)d_in[1];   // [N,512]
  const float* te_w[4] = {(const float*)d_in[2], (const float*)d_in[3],
                          (const float*)d_in[4], (const float*)d_in[5]};
  const float* te_b[4] = {(const float*)d_in[6], (const float*)d_in[7],
                          (const float*)d_in[8], (const float*)d_in[9]};
  const float* td_w[4] = {(const float*)d_in[10], (const float*)d_in[11],
                          (const float*)d_in[12], (const float*)d_in[13]};
  const float* td_b[4] = {(const float*)d_in[14], (const float*)d_in[15],
                          (const float*)d_in[16], (const float*)d_in[17]};
  const float* ve_w[4] = {(const float*)d_in[18], (const float*)d_in[19],
                          (const float*)d_in[20], (const float*)d_in[21]};
  const float* ve_b[4] = {(const float*)d_in[22], (const float*)d_in[23],
                          (const float*)d_in[24], (const float*)d_in[25]};
  const float* vd_w[4] = {(const float*)d_in[26], (const float*)d_in[27],
                          (const float*)d_in[28], (const float*)d_in[29]};
  const float* vd_b[4] = {(const float*)d_in[30], (const float*)d_in[31],
                          (const float*)d_in[32], (const float*)d_in[33]};
  const float* lora_a[4] = {(const float*)d_in[34], (const float*)d_in[35],
                            (const float*)d_in[36], (const float*)d_in[37]};
  const float* text_b[4] = {(const float*)d_in[38], (const float*)d_in[39],
                            (const float*)d_in[40], (const float*)d_in[41]};
  const float* vis_b[4]  = {(const float*)d_in[42], (const float*)d_in[43],
                            (const float*)d_in[44], (const float*)d_in[45]};
  const float* tfA = (const float*)d_in[46];  // [32,64]
  const float* tfB = (const float*)d_in[47];  // [64,32]
  const float* vfA = (const float*)d_in[48];  // [32,64]
  const float* vfB = (const float*)d_in[49];  // [64,32]

  // -- outputs --
  float* out = (float*)d_out;
  float* out_text   = out;                              // [N,768]
  float* out_vis    = out_text + (size_t)N * 768;       // [N,512]
  float* loss_t_out = out_vis + (size_t)N * 512;        // scalar
  float* loss_v_out = loss_t_out + 1;                   // scalar
  float* idx_t      = loss_v_out + 1;                   // [N,4] as float
  float* idx_v      = idx_t + (size_t)N * 4;            // [N,4] as float

  // -- workspace allocator --
  char* ws = (char*)d_ws;
  size_t woff = 0;
  auto alloc = [&](size_t bytes) -> char* {
    char* p = ws + woff;
    woff += (bytes + 255) & ~(size_t)255;
    return p;
  };

  float* h1   = (float*)alloc((size_t)N * 512 * 4);  // shared across phases
  float* h2   = (float*)alloc((size_t)N * 256 * 4);
  float* h3   = (float*)alloc((size_t)N * 128 * 4);
  float* z_t  = (float*)alloc((size_t)N * 64 * 4);
  float* z_v  = (float*)alloc((size_t)N * 64 * 4);
  float* t_res = (float*)alloc((size_t)N * 64 * 4);
  float* v_res = (float*)alloc((size_t)N * 64 * 4);
  float* t_xq  = (float*)alloc((size_t)N * 64 * 4);
  float* v_xq  = (float*)alloc((size_t)N * 64 * 4);
  float* loss_partial = (float*)alloc(2048 * 4);  // [2 mod][4 stage][256 blk]

  // packed bf16 transposed weights
  struct WSpec { const float* w; int K; int Nc; __bf16* dst; };
  WSpec specs[16] = {
      {te_w[0], 768, 512, nullptr}, {te_w[1], 512, 256, nullptr},
      {te_w[2], 256, 128, nullptr}, {te_w[3], 128, 64, nullptr},
      {td_w[0], 64, 128, nullptr},  {td_w[1], 128, 256, nullptr},
      {td_w[2], 256, 512, nullptr}, {td_w[3], 512, 768, nullptr},
      {ve_w[0], 512, 512, nullptr}, {ve_w[1], 512, 256, nullptr},
      {ve_w[2], 256, 128, nullptr}, {ve_w[3], 128, 64, nullptr},
      {vd_w[0], 64, 128, nullptr},  {vd_w[1], 128, 256, nullptr},
      {vd_w[2], 256, 512, nullptr}, {vd_w[3], 512, 512, nullptr}};
  for (int i = 0; i < 16; ++i)
    specs[i].dst = (__bf16*)alloc((size_t)specs[i].K * specs[i].Nc * 2);

  float*  cbF[8]; __bf16* cbB[8]; float* cbN[8];
  for (int i = 0; i < 8; ++i) {
    cbF[i] = (float*)alloc(256 * 64 * 4);
    cbB[i] = (__bf16*)alloc(256 * 64 * 2);
    cbN[i] = (float*)alloc(256 * 4);
  }

  // -- 1. pack weights --
  for (int i = 0; i < 16; ++i) {
    int total = specs[i].K * specs[i].Nc;
    rqvae_pack_w_kernel<<<(total + 255) / 256, 256, 0, stream>>>(
        specs[i].w, specs[i].dst, specs[i].K, specs[i].Nc);
  }

  auto gemm = [&](const float* A, const __bf16* Wt, const float* b, float* C,
                  int K, int Nc, int relu) {
    dim3 grid(Nc / 32, N / 128);
    size_t lds = (size_t)K * 64;  // 32 cols * K * 2B
    rqvae_gemm_kernel<<<grid, 256, lds, stream>>>(A, Wt, b, C, N, K, Nc, relu);
  };

  // -- 2. text encoder: 768 -> 512 -> 256 -> 128 -> 64 --
  gemm(x_text, specs[0].dst, te_b[0], h1, 768, 512, 1);
  gemm(h1, specs[1].dst, te_b[1], h2, 512, 256, 1);
  gemm(h2, specs[2].dst, te_b[2], h3, 256, 128, 1);
  gemm(h3, specs[3].dst, te_b[3], z_t, 128, 64, 0);

  // -- 3. vis encoder: 512 -> 512 -> 256 -> 128 -> 64 (reuses h1..h3) --
  gemm(x_vis, specs[8].dst, ve_b[0], h1, 512, 512, 1);
  gemm(h1, specs[9].dst, ve_b[1], h2, 512, 256, 1);
  gemm(h2, specs[10].dst, ve_b[2], h3, 256, 128, 1);
  gemm(h3, specs[11].dst, ve_b[3], z_v, 128, 64, 0);

  // -- 4. cross-modal LoRA fusion -> residuals, zero x_q --
  rqvae_fusion_kernel<<<N / 256, 256, 0, stream>>>(
      z_t, z_v, tfA, tfB, vfA, vfB, t_res, v_res, t_xq, v_xq);

  // -- 5. codebooks (low-rank A @ B per stage per modality) --
  for (int s = 0; s < 4; ++s) {
    rqvae_codebook_kernel<<<1, 256, 0, stream>>>(lora_a[s], text_b[s],
                                                 cbF[s], cbB[s], cbN[s]);
    rqvae_codebook_kernel<<<1, 256, 0, stream>>>(lora_a[s], vis_b[s],
                                                 cbF[4 + s], cbB[4 + s], cbN[4 + s]);
  }

  // -- 6. residual VQ stages (sequential per modality) --
  for (int s = 0; s < 4; ++s)
    rqvae_rq_stage_kernel<<<N / 128, 256, 0, stream>>>(
        t_res, t_xq, cbF[s], cbB[s], cbN[s], idx_t, s,
        loss_partial + s * 256);
  for (int s = 0; s < 4; ++s)
    rqvae_rq_stage_kernel<<<N / 128, 256, 0, stream>>>(
        v_res, v_xq, cbF[4 + s], cbB[4 + s], cbN[4 + s], idx_v, s,
        loss_partial + 1024 + s * 256);

  // -- 7. decoders --
  gemm(t_xq, specs[4].dst, td_b[0], h3, 64, 128, 1);
  gemm(h3, specs[5].dst, td_b[1], h2, 128, 256, 1);
  gemm(h2, specs[6].dst, td_b[2], h1, 256, 512, 1);
  gemm(h1, specs[7].dst, td_b[3], out_text, 512, 768, 0);

  gemm(v_xq, specs[12].dst, vd_b[0], h3, 64, 128, 1);
  gemm(h3, specs[13].dst, vd_b[1], h2, 128, 256, 1);
  gemm(h2, specs[14].dst, vd_b[2], h1, 256, 512, 1);
  gemm(h1, specs[15].dst, vd_b[3], out_vis, 512, 512, 0);

  // -- 8. losses --
  rqvae_finalize_kernel<<<1, 32, 0, stream>>>(loss_partial, loss_t_out, loss_v_out);
}
